// SymTripLoss_21698174779732
// MI455X (gfx1250) — matrix operand order repaired and checked by reading, not compiled, not run on hardware
//
#include <hip/hip_runtime.h>

// ---- CDNA5 vector types ----------------------------------------------------
typedef _Float16 v16h __attribute__((ext_vector_type(16)));
typedef _Float16 v8h  __attribute__((ext_vector_type(8)));
typedef _Float16 v4h  __attribute__((ext_vector_type(4)));
typedef float    v8f  __attribute__((ext_vector_type(8)));
typedef float    v4f  __attribute__((ext_vector_type(4)));

#define D       128              // embedding dim (reference constant)
#define RS      136              // halfs per LDS row: 128 + 8 pad (272 B) -> conflict-free b128 frag reads
#define TILE    16               // triplets per tile (one WMMA N dimension)

// A-operand fragment: 16x32 f16, lane l holds row l%16,
// halfs [32s + 8*(l/16) .. +7] in elems 0-7 and [.. +16 ..] in elems 8-15 (ISA 7.12.2).
__device__ __forceinline__ v16h load_fragA(const _Float16* mat, int lane, int s) {
    const _Float16* row = mat + (lane & 15) * RS;
    const int off = s * 32 + ((lane >> 4) << 3);
    v8h lo = *(const v8h*)(row + off);
    v8h hi = *(const v8h*)(row + off + 16);
    v16h r;
#pragma unroll
    for (int i = 0; i < 8; ++i) { r[i] = lo[i]; r[i + 8] = hi[i]; }
    return r;
}

// B-operand fragment: 32x16 f16 = transpose of the same row-major matrix.
// Lane l holds column l%16 (= row l%16 of X), halfs [32s + 8*(l/16) .. +7] in elems 0-7 and [.. +16 ..] in elems 8-15.
__device__ __forceinline__ v16h load_fragB(const _Float16* mat, int lane, int s) {
    const _Float16* row = mat + (lane & 15) * RS;
    const int off = s * 32 + ((lane >> 4) << 3);
    v8h lo = *(const v8h*)(row + off);
    v8h hi = *(const v8h*)(row + off + 16);
    v16h r;
#pragma unroll
    for (int i = 0; i < 8; ++i) { r[i] = lo[i]; r[i + 8] = hi[i]; }
    return r;
}

// One wave32 per tile of 16 triplets.
__global__ __launch_bounds__(32)
void symtrip_wmma_kernel(const float* __restrict__ inputs,
                         const int*   __restrict__ T,
                         float*       __restrict__ partials,
                         int n)
{
    __shared__ __align__(16) _Float16 diffs[3 * TILE * RS];  // dTA | dIA | dIT, row-major f16
    __shared__ float diagbuf[2 * 16 * 17];                   // padded 16x16 C stage (pos, neg)

    const int lane = threadIdx.x;           // 0..31, wave32
    const int tile = blockIdx.x;
    const int t0   = tile * TILE;

    const int* Ta = T;
    const int* Tt = T + n;
    const int* Ti = T + 2 * n;

    _Float16* mTA = diffs;
    _Float16* mIA = diffs + TILE * RS;
    _Float16* mIT = diffs + 2 * TILE * RS;

    // ---- Stage: coalesced row gathers (wave32 x float4 = one 512B row per load),
    //      form diffs in f32, convert to f16, store to LDS.
#pragma unroll 4
    for (int r = 0; r < TILE; ++r) {
        const int t    = t0 + r;
        const bool ok  = (t < n);
        const int ia   = ok ? Ta[t] : 0;
        const int it   = ok ? Tt[t] : 0;
        const int ii   = ok ? Ti[t] : 0;
        const float vf = ok ? 1.0f : 0.0f;

        v4f a  = *(const v4f*)(inputs + (size_t)ia * D + lane * 4);
        v4f tt = *(const v4f*)(inputs + (size_t)it * D + lane * 4);
        v4f im = *(const v4f*)(inputs + (size_t)ii * D + lane * 4);

        v4f d1 = (tt - a)  * vf;   // t - a
        v4f d2 = (im - a)  * vf;   // i - a
        v4f d3 = (im - tt) * vf;   // i - t

        v4h h1, h2, h3;
#pragma unroll
        for (int k = 0; k < 4; ++k) {
            h1[k] = (_Float16)d1[k];
            h2[k] = (_Float16)d2[k];
            h3[k] = (_Float16)d3[k];
        }
        const int ro = r * RS + lane * 4;    // 8B-aligned half offset
        *(v4h*)(mTA + ro) = h1;
        *(v4h*)(mIA + ro) = h2;
        *(v4h*)(mIT + ro) = h3;
    }
    __syncthreads();

    // ---- WMMA: diag(X * X^T) gives the squared row norms = squared distances.
    // cpos accumulates ||t-a||^2; cneg accumulates ||i-a||^2 + ||i-t||^2 (C-chaining).
    v8f cpos = {};
    v8f cneg = {};
#pragma unroll
    for (int s = 0; s < 4; ++s) {
        v16h fa = load_fragA(mTA, lane, s);
        v16h fb = load_fragB(mTA, lane, s);
        cpos = __builtin_amdgcn_wmma_f32_16x16x32_f16(false, fa, false, fb,
                                                      (short)0, cpos, false, false);
    }
#pragma unroll
    for (int s = 0; s < 4; ++s) {
        v16h fa = load_fragA(mIA, lane, s);
        v16h fb = load_fragB(mIA, lane, s);
        cneg = __builtin_amdgcn_wmma_f32_16x16x32_f16(false, fa, false, fb,
                                                      (short)0, cneg, false, false);
    }
#pragma unroll
    for (int s = 0; s < 4; ++s) {
        v16h fa = load_fragA(mIT, lane, s);
        v16h fb = load_fragB(mIT, lane, s);
        cneg = __builtin_amdgcn_wmma_f32_16x16x32_f16(false, fa, false, fb,
                                                      (short)0, cneg, false, false);
    }

    // ---- Diagonal extraction: C layout = lane n%16 holds column n, VGPR m = row mbase+m.
    const int col   = lane & 15;
    const int mbase = (lane >> 4) * 8;
    float* dp = diagbuf;
    float* dq = diagbuf + 16 * 17;
#pragma unroll
    for (int m = 0; m < 8; ++m) {
        dp[(mbase + m) * 17 + col] = cpos[m];
        dq[(mbase + m) * 17 + col] = cneg[m];
    }
    __syncthreads();

    // ---- Per-triplet loss on lanes 0..15, then wave32 reduction.
    float loss = 0.0f;
    if (lane < 16 && (t0 + lane) < n) {
        const float pos = dp[lane * 17 + lane];
        const float q   = 0.5f * dq[lane * 17 + lane];
        const float m   = fmaxf(-pos, -q);
        const float lse = m + __logf(__expf(-pos - m) + __expf(-q - m));
        loss = pos + lse;
    }
#pragma unroll
    for (int o = 16; o >= 1; o >>= 1) loss += __shfl_xor(loss, o, 32);

    if (lane == 0) partials[tile] = loss;
}

// Deterministic final reduction: fixed strided order + LDS tree, single block.
__global__ __launch_bounds__(256)
void symtrip_finalize_kernel(const float* __restrict__ partials, int nparts,
                             float* __restrict__ out, float inv_n)
{
    __shared__ float smem[256];
    float acc = 0.0f;
    for (int i = threadIdx.x; i < nparts; i += 256) acc += partials[i];
    smem[threadIdx.x] = acc;
    __syncthreads();
#pragma unroll
    for (int s = 128; s > 0; s >>= 1) {
        if (threadIdx.x < s) smem[threadIdx.x] += smem[threadIdx.x + s];
        __syncthreads();
    }
    if (threadIdx.x == 0) out[0] = smem[0] * inv_n;
}

extern "C" void kernel_launch(void* const* d_in, const int* in_sizes, int n_in,
                              void* d_out, int out_size, void* d_ws, size_t ws_size,
                              hipStream_t stream)
{
    const float* inputs = (const float*)d_in[0];   // [N_EMB, 128] f32
    // d_in[1] = targets (unused by the reference computation)
    const int*   T      = (const int*)d_in[2];     // [3, N] int32, row-major

    const int n     = in_sizes[2] / 3;
    const int tiles = (n + TILE - 1) / TILE;       // 31250 for n = 500000

    float* partials = (float*)d_ws;                // tiles * 4 bytes of scratch

    symtrip_wmma_kernel<<<tiles, 32, 0, stream>>>(inputs, T, partials, n);
    symtrip_finalize_kernel<<<1, 256, 0, stream>>>(partials, tiles, (float*)d_out,
                                                   1.0f / (float)n);
}
